// RQAutoEncoder_29463475650669
// MI455X (gfx1250) — matrix operand order repaired and checked by reading, not compiled, not run on hardware
//
#include <hip/hip_runtime.h>

// ---------------------------------------------------------------------------
// RQ-VAE forward for MI455X (gfx1250, wave32).
// Split-bf16 (hi+lo) fp32-accurate GEMMs on V_WMMA_F32_16X16X32_BF16:
//   A@W ~= Ahi@Whi + Ahi@Wlo + Alo@Whi   (lo*lo dropped, ~1e-5 rel error)
// All fp32->bf16 splitting is done ONCE (weight prep pass / producer epilogue),
// so the GEMM hot loop is pure data movement + WMMA:
//   - W: pre-transposed bf16 planes, b128 copy global->LDS (8-wave reuse)
//   - A: row-major bf16 planes, fragments loaded b128 straight from global
//   - WMMAs interleaved across N-tiles so adjacent WMMAs are independent
// VQ stages: fp32 VALU + LDS-resident codebook (2% of FLOPs).
// ---------------------------------------------------------------------------

typedef __attribute__((ext_vector_type(16))) __bf16 v16bf;
typedef __attribute__((ext_vector_type(8)))  float  v8f;
typedef unsigned short ushort_t;

#define BM  128
#define BK  32
#define WST 40   // LDS W row stride in bf16 elems: 80B (16B-aligned), bank-spread

__device__ __forceinline__ unsigned f2bf(float f) {   // fp32 -> bf16 bits, RNE
    unsigned u = __float_as_uint(f);
    return (u + 0x7FFFu + ((u >> 16) & 1u)) >> 16;
}
__device__ __forceinline__ float bf2f(unsigned b) {
    return __uint_as_float(b << 16);
}
__device__ __forceinline__ void split2(float x, unsigned& h, unsigned& l) {
    h = f2bf(x);
    l = f2bf(x - bf2f(h));
}

union Frag16 {            // one 16x16x32 bf16 operand: 16 bf16 per lane
    v16bf  v;
    float4 f4[2];         // two 16B-aligned b128 loads
};

// ---------------------------------------------------------------------------
// One-shot preps
// ---------------------------------------------------------------------------
__global__ __launch_bounds__(256)
void weight_prep(const float* __restrict__ W, ushort_t* __restrict__ WH,
                 ushort_t* __restrict__ WL, int K, int Nout)
{
    int id = blockIdx.x * 256 + threadIdx.x;          // id = k*Nout + n
    if (id < K * Nout) {
        int k = id / Nout, n = id % Nout;
        unsigned h, l;
        split2(W[id], h, l);
        WH[(size_t)n * K + k] = (ushort_t)h;          // transposed planes
        WL[(size_t)n * K + k] = (ushort_t)l;
    }
}

__global__ __launch_bounds__(256)
void act_split(const float* __restrict__ A, ushort_t* __restrict__ H,
               ushort_t* __restrict__ L, long n4)
{
    long id = (long)blockIdx.x * 256 + threadIdx.x;   // float4 id
    if (id < n4) {
        float4 v = reinterpret_cast<const float4*>(A)[id];
        unsigned h0,l0,h1,l1,h2,l2,h3,l3;
        split2(v.x,h0,l0); split2(v.y,h1,l1); split2(v.z,h2,l2); split2(v.w,h3,l3);
        reinterpret_cast<uint2*>(H)[id] = make_uint2(h0 | (h1<<16), h2 | (h3<<16));
        reinterpret_cast<uint2*>(L)[id] = make_uint2(l0 | (l1<<16), l2 | (l3<<16));
    }
}

// ---------------------------------------------------------------------------
// Split-bf16 GEMM: C = act(A @ W + b)
// MODE 0: ReLU, write bf16 hi/lo planes (intermediate layers)
// MODE 1: ReLU, write fp32            (Z_enc)
// MODE 2: no act, write fp32          (X_recon)
// ---------------------------------------------------------------------------
template<int BN, int MODE>
__global__ __launch_bounds__(256)
void gemm_bias_act(const ushort_t* __restrict__ AH, const ushort_t* __restrict__ AL,
                   const ushort_t* __restrict__ WH, const ushort_t* __restrict__ WL,
                   const float* __restrict__ bias,
                   float* __restrict__ Cf,
                   ushort_t* __restrict__ OH, ushort_t* __restrict__ OL,
                   int K, int Nout)
{
    constexpr int NT = BN / 16;                 // C tiles per wave along N
    __shared__ ushort_t WsH[BN * WST];          // W tile (already n-major/k-contig)
    __shared__ ushort_t WsL[BN * WST];

    const int t    = threadIdx.x;
    const int wave = t >> 5;                    // 8 waves -> 8 M-tiles of 16
    const int lane = t & 31;
    const int ln   = lane & 15;                 // M (A frag) / N (B frag) index
    const int h2   = lane >> 4;                 // lane-half select
    const int n0   = blockIdx.x * BN;           // n-tile fastest -> A-panel L2 reuse
    const int row0 = blockIdx.y * BM;

    v8f acc[NT];
#pragma unroll
    for (int i = 0; i < NT; ++i) acc[i] = (v8f)0.0f;

    // per-lane A fragment base (16-bit A layout: VGPR0-3 = K 8h..8h+7,
    // VGPR4-7 = K 16+8h..16+8h+7, per lane row m)
    const ushort_t* aHp = AH + (size_t)(row0 + wave * 16 + ln) * K + 8 * h2;
    const ushort_t* aLp = AL + (size_t)(row0 + wave * 16 + ln) * K + 8 * h2;

    // W-fill indexing (branchless for BN=64; half the threads for BN=32)
    const int fn = t >> 2, fpart = t & 3;       // 4 x 16B per W row

    for (int k0 = 0; k0 < K; k0 += BK) {
        // ---- W tile fill: pure b128 copy (planes are pre-split/transposed) ----
        if (BN == 64 || t < BN * 4) {
            size_t src = (size_t)(n0 + fn) * K + k0 + fpart * 8;
            *reinterpret_cast<float4*>(&WsH[fn * WST + fpart * 8]) =
                *reinterpret_cast<const float4*>(WH + src);
            *reinterpret_cast<float4*>(&WsL[fn * WST + fpart * 8]) =
                *reinterpret_cast<const float4*>(WL + src);
        }
        __syncthreads();

        // ---- A fragments straight from global bf16 planes ----
        Frag16 ah, al;
        ah.f4[0] = *reinterpret_cast<const float4*>(aHp + k0);
        ah.f4[1] = *reinterpret_cast<const float4*>(aHp + k0 + 16);
        al.f4[0] = *reinterpret_cast<const float4*>(aLp + k0);
        al.f4[1] = *reinterpret_cast<const float4*>(aLp + k0 + 16);

        // ---- B fragments for all N-tiles (lane half h -> K 8h..8h+7, 16+8h..16+8h+7) ----
        Frag16 bh[NT], bl[NT];
#pragma unroll
        for (int nt = 0; nt < NT; ++nt) {
            const ushort_t* wrH = &WsH[(nt * 16 + ln) * WST + 8 * h2];
            const ushort_t* wrL = &WsL[(nt * 16 + ln) * WST + 8 * h2];
            bh[nt].f4[0] = *reinterpret_cast<const float4*>(wrH);
            bh[nt].f4[1] = *reinterpret_cast<const float4*>(wrH + 16);
            bl[nt].f4[0] = *reinterpret_cast<const float4*>(wrL);
            bl[nt].f4[1] = *reinterpret_cast<const float4*>(wrL + 16);
        }

        // ---- WMMAs interleaved across tiles: adjacent ops are independent ----
#pragma unroll
        for (int nt = 0; nt < NT; ++nt)
            acc[nt] = __builtin_amdgcn_wmma_f32_16x16x32_bf16(
                false, ah.v, false, bh[nt].v, (short)0, acc[nt], false, false);
#pragma unroll
        for (int nt = 0; nt < NT; ++nt)
            acc[nt] = __builtin_amdgcn_wmma_f32_16x16x32_bf16(
                false, ah.v, false, bl[nt].v, (short)0, acc[nt], false, false);
#pragma unroll
        for (int nt = 0; nt < NT; ++nt)
            acc[nt] = __builtin_amdgcn_wmma_f32_16x16x32_bf16(
                false, al.v, false, bh[nt].v, (short)0, acc[nt], false, false);
        __syncthreads();
    }

    // ---- epilogue: bias (+ act), store (C layout: VGPR r -> M=r / r+8) ----
#pragma unroll
    for (int nt = 0; nt < NT; ++nt) {
        int ncol = n0 + nt * 16 + ln;
        float bv = bias[ncol];
#pragma unroll
        for (int r = 0; r < 8; ++r) {
            float v = acc[nt][r] + bv;
            if (MODE != 2) v = fmaxf(v, 0.0f);
            int rowm = row0 + wave * 16 + r + (h2 << 3);
            size_t o = (size_t)rowm * Nout + ncol;
            if (MODE == 0) {
                unsigned h, l;
                split2(v, h, l);
                OH[o] = (ushort_t)h;
                OL[o] = (ushort_t)l;
            } else {
                Cf[o] = v;
            }
        }
    }
}

// ---------------------------------------------------------------------------
// One residual-VQ stage (fp32 exact): argmin_c (0.5*|cb[c]|^2 - q.cb[c])
// Optionally emits split-bf16 planes of the residual (for the decoder input).
// ---------------------------------------------------------------------------
__global__ __launch_bounds__(256)
void vq_stage(const float* __restrict__ q, const float* __restrict__ cb,
              float* __restrict__ dec, float* __restrict__ res,
              int* __restrict__ idx,
              ushort_t* __restrict__ resH, ushort_t* __restrict__ resL)
{
    __shared__ float cbs[256 * 32];
    __shared__ float hn[256];

    const int t = threadIdx.x;
#pragma unroll
    for (int i = 0; i < 8; ++i) {
        int id = t + 256 * i;
        reinterpret_cast<float4*>(cbs)[id] =
            reinterpret_cast<const float4*>(cb)[id];
    }
    __syncthreads();
    {
        const float* c = cbs + t * 32;
        float s = 0.0f;
#pragma unroll
        for (int d = 0; d < 32; ++d) s = fmaf(c[d], c[d], s);
        hn[t] = 0.5f * s;
    }
    __syncthreads();

    const int row = blockIdx.x * 256 + t;
    float qv[32];
    const float4* q4 = reinterpret_cast<const float4*>(q + (size_t)row * 32);
#pragma unroll
    for (int i = 0; i < 8; ++i) {
        float4 v = q4[i];
        qv[4 * i + 0] = v.x; qv[4 * i + 1] = v.y;
        qv[4 * i + 2] = v.z; qv[4 * i + 3] = v.w;
    }

    float best = __builtin_inff();
    int bi = 0;
    for (int c = 0; c < 256; ++c) {          // uniform LDS addr -> broadcast
        const float* cc = cbs + c * 32;
        float dot = 0.0f;
#pragma unroll
        for (int d = 0; d < 32; ++d) dot = fmaf(qv[d], cc[d], dot);
        float sc = hn[c] - dot;
        if (sc < best) { best = sc; bi = c; }  // strict < : first-min tie-break
    }

    const float* cc = cbs + bi * 32;
    float rv[32];
#pragma unroll
    for (int d = 0; d < 32; ++d) rv[d] = qv[d] - cc[d];

    float4* d4 = reinterpret_cast<float4*>(dec + (size_t)row * 32);
    float4* r4 = reinterpret_cast<float4*>(res + (size_t)row * 32);
#pragma unroll
    for (int i = 0; i < 8; ++i) {
        d4[i] = make_float4(cc[4*i], cc[4*i+1], cc[4*i+2], cc[4*i+3]);
        r4[i] = make_float4(rv[4*i], rv[4*i+1], rv[4*i+2], rv[4*i+3]);
    }
    idx[row] = bi;

    if (resH) {                              // split planes for decoder input
        uint2* h2o = reinterpret_cast<uint2*>(resH + (size_t)row * 32);
        uint2* l2o = reinterpret_cast<uint2*>(resL + (size_t)row * 32);
#pragma unroll
        for (int i = 0; i < 8; ++i) {
            unsigned h0,l0,h1,l1,h2,l2,h3,l3;
            split2(rv[4*i+0],h0,l0); split2(rv[4*i+1],h1,l1);
            split2(rv[4*i+2],h2,l2); split2(rv[4*i+3],h3,l3);
            h2o[i] = make_uint2(h0 | (h1<<16), h2 | (h3<<16));
            l2o[i] = make_uint2(l0 | (l1<<16), l2 | (l3<<16));
        }
    }
}

// ---------------------------------------------------------------------------
extern "C" void kernel_launch(void* const* d_in, const int* in_sizes, int n_in,
                              void* d_out, int out_size, void* d_ws, size_t ws_size,
                              hipStream_t stream)
{
    const int N = in_sizes[0] / 768;

    const float* X   = (const float*)d_in[0];
    const float* Wf[8] = { (const float*)d_in[1],  (const float*)d_in[3],
                           (const float*)d_in[5],  (const float*)d_in[7],
                           (const float*)d_in[12], (const float*)d_in[14],
                           (const float*)d_in[16], (const float*)d_in[18] };
    const float* Bf[8] = { (const float*)d_in[2],  (const float*)d_in[4],
                           (const float*)d_in[6],  (const float*)d_in[8],
                           (const float*)d_in[13], (const float*)d_in[15],
                           (const float*)d_in[17], (const float*)d_in[19] };
    const float* cb1 = (const float*)d_in[9];
    const float* cb2 = (const float*)d_in[10];
    // d_in[11] == cb3: unused (reference reuses cb2 for stage 3 — faithful)

    const int LK[8] = {768, 512, 256, 128,  32, 128, 256, 512};
    const int LN[8] = {512, 256, 128,  32, 128, 256, 512, 768};

    float* out = (float*)d_out;
    const size_t nz = (size_t)N * 32;
    float* Z_enc  = out + 0 * nz;
    float* Z_dec1 = out + 1 * nz;
    float* z_res1 = out + 2 * nz;
    float* z_dec2 = out + 3 * nz;
    float* z_res2 = out + 4 * nz;
    float* z_dec3 = out + 5 * nz;
    float* z_res3 = out + 6 * nz;
    float* Xrec   = out + 7 * nz;
    int*   idx1   = (int*)(out + 7 * nz + (size_t)N * 768);
    int*   idx2   = idx1 + N;
    int*   idx3   = idx2 + N;

    // ---- workspace carve-up ----
    unsigned char* p = (unsigned char*)d_ws;
    auto carve = [&p](size_t bytes) {
        void* r = (void*)p;
        p += (bytes + 255) & ~(size_t)255;
        return r;
    };
    ushort_t *WHp[8], *WLp[8];
    for (int l = 0; l < 8; ++l) {
        size_t e = (size_t)LK[l] * LN[l] * sizeof(ushort_t);
        WHp[l] = (ushort_t*)carve(e);
        WLp[l] = (ushort_t*)carve(e);
    }
    ushort_t* XH  = (ushort_t*)carve((size_t)N * 768 * 2);
    ushort_t* XL  = (ushort_t*)carve((size_t)N * 768 * 2);
    ushort_t* sAH = (ushort_t*)carve((size_t)N * 512 * 2);   // ping (<=512 wide)
    ushort_t* sAL = (ushort_t*)carve((size_t)N * 512 * 2);
    ushort_t* sBH = (ushort_t*)carve((size_t)N * 256 * 2);   // pong (<=256 wide)
    ushort_t* sBL = (ushort_t*)carve((size_t)N * 256 * 2);
    ushort_t* rH  = (ushort_t*)carve((size_t)N * 32 * 2);
    ushort_t* rL  = (ushort_t*)carve((size_t)N * 32 * 2);

    dim3 blk(256);
    const int MB = N / BM;

    // ---- one-shot preps: split weights (transposed planes) and X ----
    for (int l = 0; l < 8; ++l) {
        int e = LK[l] * LN[l];
        weight_prep<<<dim3((e + 255) / 256), blk, 0, stream>>>(
            Wf[l], WHp[l], WLp[l], LK[l], LN[l]);
    }
    {
        long n4 = (long)N * 768 / 4;
        act_split<<<dim3((unsigned)((n4 + 255) / 256)), blk, 0, stream>>>(X, XH, XL, n4);
    }

    // ---- encoder: 768 -> 512 -> 256 -> 128 -> 32 (ReLU each) ----
    gemm_bias_act<64, 0><<<dim3(8, MB), blk, 0, stream>>>(
        XH, XL, WHp[0], WLp[0], Bf[0], nullptr, sAH, sAL, 768, 512);
    gemm_bias_act<64, 0><<<dim3(4, MB), blk, 0, stream>>>(
        sAH, sAL, WHp[1], WLp[1], Bf[1], nullptr, sBH, sBL, 512, 256);
    gemm_bias_act<64, 0><<<dim3(2, MB), blk, 0, stream>>>(
        sBH, sBL, WHp[2], WLp[2], Bf[2], nullptr, sAH, sAL, 256, 128);
    gemm_bias_act<32, 1><<<dim3(1, MB), blk, 0, stream>>>(
        sAH, sAL, WHp[3], WLp[3], Bf[3], Z_enc, nullptr, nullptr, 128, 32);

    // ---- residual VQ (cb1, cb2, cb2) ----
    dim3 vgrid(N / 256);
    vq_stage<<<vgrid, blk, 0, stream>>>(Z_enc,  cb1, Z_dec1, z_res1, idx1, nullptr, nullptr);
    vq_stage<<<vgrid, blk, 0, stream>>>(z_res1, cb2, z_dec2, z_res2, idx2, nullptr, nullptr);
    vq_stage<<<vgrid, blk, 0, stream>>>(z_res2, cb2, z_dec3, z_res3, idx3, rH, rL);

    // ---- decoder: 32 -> 128 -> 256 -> 512 -> 768 (last layer: no ReLU) ----
    gemm_bias_act<64, 0><<<dim3(2,  MB), blk, 0, stream>>>(
        rH, rL, WHp[4], WLp[4], Bf[4], nullptr, sAH, sAL, 32, 128);
    gemm_bias_act<64, 0><<<dim3(4,  MB), blk, 0, stream>>>(
        sAH, sAL, WHp[5], WLp[5], Bf[5], nullptr, sBH, sBL, 128, 256);
    gemm_bias_act<64, 0><<<dim3(8,  MB), blk, 0, stream>>>(
        sBH, sBL, WHp[6], WLp[6], Bf[6], nullptr, sAH, sAL, 256, 512);
    gemm_bias_act<64, 2><<<dim3(12, MB), blk, 0, stream>>>(
        sAH, sAL, WHp[7], WLp[7], Bf[7], Xrec, nullptr, nullptr, 512, 768);
}
